// BinarizedConvBlock_120259085187
// MI455X (gfx1250) — matrix operand order repaired and checked by reading, not compiled, not run on hardware
//
#include <hip/hip_runtime.h>
#include <hip/hip_bf16.h>
#include <stdint.h>

typedef __attribute__((ext_vector_type(16))) _Float16 v16h;
typedef __attribute__((ext_vector_type(8)))  _Float16 v8h;
typedef __attribute__((ext_vector_type(8)))  float    v8f;

union AB16 { v16h v; v8h h[2]; };

// ---------------- problem constants ----------------
#define CIN   256
#define COUT  256
#define BATCH 32
#define H     56
#define W_IMG 56
#define KDIM  2304        // 9 * CIN, K order = (tap, ci), tap = kh*3+kw
#define HW    3136        // 56*56
#define NTOT  100352      // BATCH * HW
#define POOLN 6422528     // 32*256*28*28

// GEMM tiling
#define TM 64             // block M tile
#define TN 128            // block N tile
#define TK 32             // K step
#define TKP 40            // padded LDS row (80B stride -> conflict-free b128)

// ---------------- kernel 1: binarize weights to f16, K-order (tap, ci) ----------------
__global__ __launch_bounds__(256) void binarize_w_kernel(
    const float* __restrict__ Wf, _Float16* __restrict__ Wb, int n) {
  int i = blockIdx.x * 256 + threadIdx.x;
  if (i >= n) return;
  float w = Wf[i];                 // [co][ci][kh][kw]
  float s = (w > 0.f) ? 1.f : ((w < 0.f) ? -1.f : 0.f);
  int co  = i / KDIM;
  int r   = i - co * KDIM;
  int ci  = r / 9;
  int tap = r - ci * 9;
  Wb[co * KDIM + tap * CIN + ci] = (_Float16)s;
}

// ---------------- kernel 2: x NCHW fp32 -> NHWC f16 (LDS-tiled transpose) ----------------
// tile = 32 channels x 32 hw-positions; 3136 = 98*32 and 256 = 8*32 (no remainders)
// grid = (98 hw-tiles, 8 c-tiles, 32 batches), block = 256 = 8x32
__global__ __launch_bounds__(256) void convert_x_nhwc_kernel(
    const float* __restrict__ x, _Float16* __restrict__ xh) {
  __shared__ _Float16 tile[32][33];
  const int tx = threadIdx.x & 31;      // fast index
  const int ty = threadIdx.x >> 5;      // 0..7
  const int hw0 = blockIdx.x * 32;
  const int c0  = blockIdx.y * 32;
  const int b   = blockIdx.z;
#pragma unroll
  for (int cc = 0; cc < 32; cc += 8) {  // coalesced read along hw
    float v = x[((b * CIN + c0 + cc + ty) * HW) + hw0 + tx];
    tile[cc + ty][tx] = (_Float16)v;
  }
  __syncthreads();
#pragma unroll
  for (int ww = 0; ww < 32; ww += 8) {  // coalesced write along c
    xh[(b * HW + hw0 + ww + ty) * CIN + c0 + tx] = tile[tx][ww + ty];
  }
}

// ---------------- kernel 3: implicit-GEMM conv via WMMA ----------------
// A = Wb [COUT][KDIM] (M x K), B = im2col(xh NHWC) (K x N), D = y (M x N) fp32
// block = 256 threads = 8 waves (2 waveM x 4 waveN), wave tile 32x32 -> 4 WMMA
// K-order (tap, ci): a lane's 16 consecutive k = 16 consecutive channels at one
// pixel = 2x global_load_b128, bounds check uniform per tap.
// A tile staged with GLOBAL_LOAD_ASYNC_TO_LDS_B128 (ASYNCcnt), ping-pong LDS,
// K-loop unrolled x2 so buffer offsets are compile-time constants.
__global__ __launch_bounds__(256) void conv_wmma_kernel(
    const _Float16* __restrict__ xh, const _Float16* __restrict__ wb,
    float* __restrict__ y) {
  __shared__ _Float16 Alds[2][TM][TKP];   // 2 x 5120 B
  __shared__ _Float16 Blds[2][TN][TKP];   // 2 x 10240 B (transposed [n][k])

  const int tid  = threadIdx.x;
  const int lane = tid & 31;
  const int wave = tid >> 5;
  const int waveM = wave >> 2;         // 0..1
  const int waveN = wave & 3;          // 0..3

  const int mBlock = blockIdx.y * TM;
  const int nBlock = blockIdx.x * TN;

  // staging index maps
  const int aRow = tid >> 2;           // 0..63
  const int aCol = (tid & 3) * 8;      // 0,8,16,24
  const int bN   = tid & 127;          // B column 0..127
  const int bK0  = (tid >> 7) * 16;    // 0 or 16

  // output-pixel decomposition for this thread's B column
  const int nIdx = nBlock + bN;
  const int bb = nIdx / HW;
  const int pp = nIdx - bb * HW;
  const int oh = pp / W_IMG;
  const int ow = pp - oh * W_IMG;

  const unsigned aGlobOff = (unsigned)(((mBlock + aRow) * KDIM + aCol) * 2);

  v8f acc00 = {}, acc01 = {}, acc10 = {}, acc11 = {};

  auto stageA = [&](int buf, int k0) {
    unsigned ldsAddr = (unsigned)(uintptr_t)&Alds[buf][aRow][aCol];
    unsigned gOff    = aGlobOff + (unsigned)(k0 * 2);
    asm volatile("global_load_async_to_lds_b128 %0, %1, %2"
                 :: "v"(ldsAddr), "v"(gOff), "s"(wb) : "memory");
  };
  auto stageB = [&](int buf, int k0) {
    const int kk  = k0 + bK0;          // multiple of 16, within one tap block
    const int tap = kk >> 8;           // 0..8
    const int ci0 = kk & 255;
    const int kh  = tap / 3;
    const int kw  = tap - kh * 3;
    const int ih  = oh - 1 + kh;
    const int iw  = ow - 1 + kw;
    v8h lo = {}, hi = {};
    if ((unsigned)ih < (unsigned)H && (unsigned)iw < (unsigned)W_IMG) {
      const _Float16* src = xh + ((bb * HW + ih * W_IMG + iw) * CIN + ci0);
      lo = *(const v8h*)src;           // 32B-aligned (ci0 multiple of 16)
      hi = *(const v8h*)(src + 8);
    }
    *(v8h*)&Blds[buf][bN][bK0]     = lo;
    *(v8h*)&Blds[buf][bN][bK0 + 8] = hi;
  };

  // fragment coordinates (ISA VGPR layouts)
  const int ar0 = waveM * 32 + (lane & 15);
  const int ak  = (lane >> 4) * 8;     // A: lanes<16 K 0-7&16-23, lanes>=16 K 8-15&24-31
  const int bc0 = waveN * 32 + (lane & 15);
  const int bk  = (lane >> 4) * 8;     // B: lanes<16 K 0-7&16-23, lanes>=16 K 8-15&24-31

  auto compute = [&](const _Float16 (*A)[TKP], const _Float16 (*B)[TKP]) {
    AB16 a0; a0.h[0] = *(const v8h*)&A[ar0][ak];
             a0.h[1] = *(const v8h*)&A[ar0][ak + 16];
    AB16 a1; a1.h[0] = *(const v8h*)&A[ar0 + 16][ak];
             a1.h[1] = *(const v8h*)&A[ar0 + 16][ak + 16];
    AB16 b0; b0.h[0] = *(const v8h*)&B[bc0][bk];
             b0.h[1] = *(const v8h*)&B[bc0][bk + 16];
    AB16 b1; b1.h[0] = *(const v8h*)&B[bc0 + 16][bk];
             b1.h[1] = *(const v8h*)&B[bc0 + 16][bk + 16];
    acc00 = __builtin_amdgcn_wmma_f32_16x16x32_f16(false, a0.v, false, b0.v,
                                                   (short)0, acc00, false, false);
    acc01 = __builtin_amdgcn_wmma_f32_16x16x32_f16(false, a0.v, false, b1.v,
                                                   (short)0, acc01, false, false);
    acc10 = __builtin_amdgcn_wmma_f32_16x16x32_f16(false, a1.v, false, b0.v,
                                                   (short)0, acc10, false, false);
    acc11 = __builtin_amdgcn_wmma_f32_16x16x32_f16(false, a1.v, false, b1.v,
                                                   (short)0, acc11, false, false);
  };

  // prologue: stage tile 0 into buffer 0
  stageA(0, 0);
  stageB(0, 0);
  asm volatile("s_wait_asynccnt 0x0" ::: "memory");
  __syncthreads();

  // KDIM/TK = 72 steps, unrolled x2 over ping-pong buffers (36 iterations)
  for (int k0 = 0; k0 < KDIM; k0 += 2 * TK) {
    // stage k0+TK into buf1 while computing buf0
    if (k0 + TK < KDIM) { stageA(1, k0 + TK); stageB(1, k0 + TK); }
    compute(Alds[0], Blds[0]);
    asm volatile("s_wait_asynccnt 0x0" ::: "memory");
    __syncthreads();

    // stage k0+2*TK into buf0 while computing buf1
    if (k0 + 2 * TK < KDIM) { stageA(0, k0 + 2 * TK); stageB(0, k0 + 2 * TK); }
    compute(Alds[1], Blds[1]);
    asm volatile("s_wait_asynccnt 0x0" ::: "memory");
    __syncthreads();
  }

  // epilogue: C/D layout -> VGPR r, lane l: M = r + 8*(l>=16), N = l&15
  const int coBase = mBlock + waveM * 32 + ((lane >> 4) * 8);
#pragma unroll
  for (int mi = 0; mi < 2; ++mi) {
#pragma unroll
    for (int nj = 0; nj < 2; ++nj) {
      int nOut = nBlock + waveN * 32 + nj * 16 + (lane & 15);
      int b2 = nOut / HW;
      int p2 = nOut - b2 * HW;
      int oh2 = p2 / W_IMG;
      int ow2 = p2 - oh2 * W_IMG;
      float* yp = y + (((b2 * COUT + coBase + mi * 16) * H) + oh2) * W_IMG + ow2;
      const v8f& acc = mi ? (nj ? acc11 : acc10) : (nj ? acc01 : acc00);
#pragma unroll
      for (int r = 0; r < 8; ++r)
        yp[r * HW] = acc[r];
    }
  }
}

// ---------------- kernel 4: per-channel batch stats -> scale/shift ----------------
__global__ __launch_bounds__(256) void bn_stats_kernel(
    const float* __restrict__ y, const float* __restrict__ gamma,
    const float* __restrict__ beta, float* __restrict__ scale,
    float* __restrict__ shift) {
  const int c = blockIdx.x;
  const int tid = threadIdx.x;
  float s = 0.f, s2 = 0.f;
  for (int i = tid; i < BATCH * HW; i += 256) {
    int b = i / HW;
    int p = i - b * HW;
    float v = y[(b * COUT + c) * HW + p];
    s += v; s2 += v * v;
  }
  __shared__ float sh[256], sh2[256];
  sh[tid] = s; sh2[tid] = s2;
  __syncthreads();
  for (int off = 128; off > 0; off >>= 1) {
    if (tid < off) { sh[tid] += sh[tid + off]; sh2[tid] += sh2[tid + off]; }
    __syncthreads();
  }
  if (tid == 0) {
    const float inv_n = 1.f / (float)(BATCH * HW);
    float mean = sh[0] * inv_n;
    float var  = sh2[0] * inv_n - mean * mean;
    float sc = gamma[c] * rsqrtf(var + 1e-5f);
    scale[c] = sc;
    shift[c] = beta[c] - mean * sc;
  }
}

// ---------------- kernel 5: BN-affine + ReLU + 2x2 maxpool ----------------
__global__ __launch_bounds__(256) void bn_relu_pool_kernel(
    const float* __restrict__ y, const float* __restrict__ scale,
    const float* __restrict__ shift, float* __restrict__ out) {
  int o = blockIdx.x * 256 + threadIdx.x;
  if (o >= POOLN) return;
  int ow = o % 28;
  int t  = o / 28;
  int oh = t % 28;  t /= 28;
  int c  = t % COUT;
  int b  = t / COUT;
  const float* base = y + ((b * COUT + c) * H + oh * 2) * W_IMG + ow * 2;
  float sc = scale[c], sf = shift[c];
  float v00 = fmaxf(fmaf(base[0],         sc, sf), 0.f);
  float v01 = fmaxf(fmaf(base[1],         sc, sf), 0.f);
  float v10 = fmaxf(fmaf(base[W_IMG],     sc, sf), 0.f);
  float v11 = fmaxf(fmaf(base[W_IMG + 1], sc, sf), 0.f);
  out[o] = fmaxf(fmaxf(v00, v01), fmaxf(v10, v11));
}

// ---------------- launcher ----------------
extern "C" void kernel_launch(void* const* d_in, const int* in_sizes, int n_in,
                              void* d_out, int out_size, void* d_ws, size_t ws_size,
                              hipStream_t stream) {
  const float* x     = (const float*)d_in[0];   // [32,256,56,56]
  const float* Wf    = (const float*)d_in[1];   // [256,256,3,3]
  const float* gamma = (const float*)d_in[2];   // [256]
  const float* beta  = (const float*)d_in[3];   // [256]
  float* out = (float*)d_out;                   // [32,256,28,28]

  // workspace layout (bytes, 16B aligned)
  char* ws = (char*)d_ws;
  const size_t off_xh    = 0;                                           // 51,380,224 B
  const size_t off_wb    = off_xh + (size_t)BATCH * CIN * HW * 2;       // +1,179,648 B
  const size_t off_y     = off_wb + (size_t)COUT * KDIM * 2;            // +102,760,448 B
  const size_t off_scale = off_y  + (size_t)BATCH * COUT * HW * 4;
  const size_t off_shift = off_scale + 1024;

  _Float16* xh    = (_Float16*)(ws + off_xh);    // NHWC f16
  _Float16* wb    = (_Float16*)(ws + off_wb);    // [co][tap*256+ci] f16 in {-1,0,1}
  float*    y     = (float*)(ws + off_y);        // NCHW f32 conv output
  float*    scale = (float*)(ws + off_scale);
  float*    shift = (float*)(ws + off_shift);

  // 1) binarize + K-reorder weights
  {
    int n = COUT * KDIM;                       // 589,824
    binarize_w_kernel<<<(n + 255) / 256, 256, 0, stream>>>(Wf, wb, n);
  }
  // 2) NCHW -> NHWC f16 transpose (LDS-tiled, fully coalesced both sides)
  {
    dim3 grid(HW / 32, CIN / 32, BATCH);       // (98, 8, 32)
    convert_x_nhwc_kernel<<<grid, 256, 0, stream>>>(x, xh);
  }
  // 3) implicit-GEMM conv
  {
    dim3 grid(NTOT / TN, COUT / TM);           // (784, 4)
    conv_wmma_kernel<<<grid, 256, 0, stream>>>(xh, wb, y);
  }
  // 4) batch-norm statistics per channel
  bn_stats_kernel<<<COUT, 256, 0, stream>>>(y, gamma, beta, scale, shift);
  // 5) fused BN + ReLU + maxpool
  bn_relu_pool_kernel<<<(POOLN + 255) / 256, 256, 0, stream>>>(y, scale, shift, out);
}
